// Attention_82016695485372
// MI455X (gfx1250) — compile-verified, matrix operand order checked by reading, not run on hardware
//
#include <hip/hip_runtime.h>
#include <hip/hip_bf16.h>

// ---------------------------------------------------------------------------
// MI455X (gfx1250) attention block.
// ~163 GFLOP vs ~100MB unique data -> matrix-core bound. All GEMMs run as
// v_wmma_f32_16x16x32_bf16 (bf16 in, f32 accum). Global->LDS staging is done
// by the Tensor Data Mover (tensor_load_to_lds, TENSORcnt), double buffered,
// so the GEMM inner loop is pure ds_load_b128 + v_wmma.
// ---------------------------------------------------------------------------

typedef __bf16 bf16_t;
typedef __attribute__((ext_vector_type(16))) __bf16 v16bf;
typedef __attribute__((ext_vector_type(8)))  __bf16 v8bf;
typedef __attribute__((ext_vector_type(8)))  float  v8f;
typedef __attribute__((ext_vector_type(4)))  unsigned int u32x4;
typedef __attribute__((ext_vector_type(8)))  int i32x8;
typedef __attribute__((ext_vector_type(4)))  int i32x4;

#define S_DIM 4096
#define N_DIM 1024
#define F_DIM 4096

__device__ __forceinline__ bf16_t f2bf(float f) {
  unsigned u = __builtin_bit_cast(unsigned, f);
  u += 0x7FFFu + ((u >> 16) & 1u);        // round-to-nearest-even
  unsigned short h = (unsigned short)(u >> 16);
  return __builtin_bit_cast(bf16_t, h);
}

// ---------------------------------------------------------------------------
// TDM: issue a 2D tensor_load_to_lds of a [128 rows x 32 cols] bf16 tile.
//   gsrc     : global address of tile start (row stride = k_stride elements)
//   lds_addr : LDS byte address of destination
// LDS padding: rows are 16 DWORDs; pad_interval=16DW (code 3), pad_amount=4DW
// (code 3) -> effective LDS row stride = 20 DW = 40 bf16 elements.
// ---------------------------------------------------------------------------
#define TDM_TILE_K 32
#define TDM_TILE_R 128

__device__ __forceinline__ void tdm_load_tile(const bf16_t* gsrc,
                                              unsigned lds_addr,
                                              unsigned k_stride) {
  const unsigned long long ga = (unsigned long long)(uintptr_t)gsrc;

  u32x4 g0;
  g0.x = 1u;                                              // count=1, user mode
  g0.y = lds_addr;                                        // lds_addr[31:0]
  g0.z = (unsigned)ga;                                    // global_addr[31:0]
  g0.w = (unsigned)((ga >> 32) & 0x01FFFFFFu) | (2u << 30); // addr[56:32]|type=2

  const unsigned td0 = 0x40000000u;   // huge tensor dims: tiles never go OOB
  const unsigned td1 = 0x40000000u;
  i32x8 g1;
  g1[0] = (int)((1u << 16)     // data_size = 1 -> 2 bytes
              | (1u << 20)     // pad_enable
              | (3u << 22)     // pad_interval code 3 -> every 16 DWORDs
              | (3u << 25));   // pad_amount  code 3 -> 4 DWORDs
  g1[1] = (int)((td0 & 0xFFFFu) << 16);                   // abar=0 | dim0 lo16
  g1[2] = (int)((td0 >> 16) | ((td1 & 0xFFFFu) << 16));   // dim0 hi | dim1 lo
  g1[3] = (int)((td1 >> 16) | ((unsigned)TDM_TILE_K << 16)); // dim1 hi|tile0
  g1[4] = (int)TDM_TILE_R;                                // tile1 | tile2=0
  g1[5] = (int)k_stride;                                  // dim0_stride lo32
  g1[6] = 0;                                              // stride hi | d1s lo
  g1[7] = 0;

  i32x4 g2 = {0, 0, 0, 0};
  i32x4 g3 = {0, 0, 0, 0};

#if __has_include(<hip/amd_detail/amd_gfx1250_TDM.h>)
  // amdgpu-toolchain (therock-10.0 headers): 6-arg builtin
  i32x8 gx = {0, 0, 0, 0, 0, 0, 0, 0};
  __builtin_amdgcn_tensor_load_to_lds(g0, g1, g2, g3, gx, 0);
#else
  // ROCm 7.2 (clang-22): 5-arg builtin
  __builtin_amdgcn_tensor_load_to_lds(g0, g1, g2, g3, 0);
#endif
}

// ---------------------------------------------------------------------------
// LayerNorm: one block per row; emits f32 (residual path) + bf16 (GEMM path).
// ---------------------------------------------------------------------------
__global__ __launch_bounds__(256)
void layernorm_kernel(const float* __restrict__ x,
                      const float* __restrict__ g,
                      const float* __restrict__ b,
                      float* __restrict__ xn,
                      bf16_t* __restrict__ xnb) {
  const int row = blockIdx.x;
  const int tid = threadIdx.x;
  const float* xr = x + (size_t)row * N_DIM;
  __shared__ float red[256];

  float lx[4];
  float s = 0.f;
#pragma unroll
  for (int i = 0; i < 4; ++i) { lx[i] = xr[tid + i * 256]; s += lx[i]; }
  red[tid] = s; __syncthreads();
  for (int st = 128; st > 0; st >>= 1) {
    if (tid < st) red[tid] += red[tid + st];
    __syncthreads();
  }
  const float mu = red[0] * (1.0f / N_DIM);
  __syncthreads();

  s = 0.f;
#pragma unroll
  for (int i = 0; i < 4; ++i) { float d = lx[i] - mu; s += d * d; }
  red[tid] = s; __syncthreads();
  for (int st = 128; st > 0; st >>= 1) {
    if (tid < st) red[tid] += red[tid + st];
    __syncthreads();
  }
  const float rstd = rsqrtf(red[0] * (1.0f / N_DIM) + 1e-5f);

#pragma unroll
  for (int i = 0; i < 4; ++i) {
    const int c = tid + i * 256;
    const float v = (lx[i] - mu) * rstd * g[c] + b[c];
    xn [(size_t)row * N_DIM + c] = v;
    xnb[(size_t)row * N_DIM + c] = f2bf(v);
  }
}

// ---------------------------------------------------------------------------
// Transpose + f32->bf16 convert: W [K,N] f32  ->  Wt [N,K] bf16.
// ---------------------------------------------------------------------------
__global__ __launch_bounds__(256)
void transpose_cvt_kernel(const float* __restrict__ W,
                          bf16_t* __restrict__ Wt,
                          int K, int N) {
  __shared__ float tile[32][33];
  const int tx = threadIdx.x & 31;
  const int ty = threadIdx.x >> 5;          // 0..7
  const int bn = blockIdx.x * 32;
  const int bk = blockIdx.y * 32;
#pragma unroll
  for (int i = ty; i < 32; i += 8)
    tile[i][tx] = W[(size_t)(bk + i) * N + (bn + tx)];
  __syncthreads();
#pragma unroll
  for (int i = ty; i < 32; i += 8)
    Wt[(size_t)(bn + i) * K + (bk + tx)] = f2bf(tile[tx][i]);
}

// ---------------------------------------------------------------------------
// WMMA GEMM: C[M,N] = A[M,K](bf16) @ Bt[N,K](bf16)^T, f32 accumulate.
// Block tile 128x128x32; 8 wave32 waves in 4(M)x2(N), each wave owns 32x64 =
// 2x4 v_wmma_f32_16x16x32_bf16 tiles. TDM double-buffered staging, prefetch
// depth 2; wave 0 drives the tensor engine + s_wait_tensorcnt.
// ---------------------------------------------------------------------------
enum {
  EPI_BIAS  = 1,   // += bias[col]
  EPI_LEAKY = 2,   // leaky_relu(0.1)
  EPI_RES   = 4,   // += res[row,col]
  EPI_F32   = 8,   // store f32 row-major
  EPI_BF16  = 16,  // store bf16 row-major
  EPI_BF16T = 32,  // store bf16 transposed (C^T), leading dim ldt
};

template <int EPI>
__global__ __launch_bounds__(256)
void gemm_wmma_kernel(const bf16_t* __restrict__ A,    // [M,K]
                      const bf16_t* __restrict__ Bt,   // [N,K]
                      const float*  __restrict__ bias, // [N] or null
                      const float*  __restrict__ res,  // [M,N] or null
                      float*  __restrict__ Cf,
                      bf16_t* __restrict__ Cb,
                      int M, int N, int K, int ldt) {
  constexpr int BM = 128, BN = 128, BK = 32, LDA = BK + 8; // stride 40 via TDM pad
  __shared__ bf16_t sA[2][BM * LDA];
  __shared__ bf16_t sB[2][BN * LDA];

  const int tid  = threadIdx.x;
  const int lane = tid & 31;
  const int wid  = tid >> 5;         // 0..7
  const int wm   = wid & 3;          // 4 waves along M (32 rows each)
  const int wn   = wid >> 2;         // 2 waves along N (64 cols each)

  const int blockN0 = blockIdx.x * BN;
  const int blockM0 = blockIdx.y * BM;

  const int r  = lane & 15;
  const int hf = lane >> 4;          // which K-half this lane holds

  const bf16_t* gA = A  + (size_t)blockM0 * K;
  const bf16_t* gB = Bt + (size_t)blockN0 * K;
  const unsigned ldsA[2] = { (unsigned)(uintptr_t)(const void*)&sA[0][0],
                             (unsigned)(uintptr_t)(const void*)&sA[1][0] };
  const unsigned ldsB[2] = { (unsigned)(uintptr_t)(const void*)&sB[0][0],
                             (unsigned)(uintptr_t)(const void*)&sB[1][0] };

  const int NK = K / BK;

  // ---- prologue: fill buf0 (+ buf1 if it exists), wait for buf0
  if (wid == 0) {
    tdm_load_tile(gA, ldsA[0], (unsigned)K);
    tdm_load_tile(gB, ldsB[0], (unsigned)K);
    if (NK > 1) {
      tdm_load_tile(gA + BK, ldsA[1], (unsigned)K);
      tdm_load_tile(gB + BK, ldsB[1], (unsigned)K);
      __builtin_amdgcn_s_wait_tensorcnt(2);   // in-order: buf0's pair done
    } else {
      __builtin_amdgcn_s_wait_tensorcnt(0);
    }
  }
  __syncthreads();

  v8f acc[2][4] = {};

  for (int kt = 0; kt < NK; ++kt) {
    const int cur = kt & 1;

    // ---- fragments per ISA 16-bit A/B layout:
    // lanes 0-15 hold K in {0..7, 16..23}; lanes 16-31 hold {8..15, 24..31}
    v16bf afr[2], bfr[4];
#pragma unroll
    for (int mt = 0; mt < 2; ++mt) {
      const bf16_t* p0 = &sA[cur][(wm * 32 + mt * 16 + r) * LDA + hf * 8];
      union { v16bf v; v8bf h[2]; } u;
      u.h[0] = *(const v8bf*)(p0);
      u.h[1] = *(const v8bf*)(p0 + 16);
      afr[mt] = u.v;
    }
#pragma unroll
    for (int nt = 0; nt < 4; ++nt) {
      const bf16_t* p0 = &sB[cur][(wn * 64 + nt * 16 + r) * LDA + hf * 8];
      union { v16bf v; v8bf h[2]; } u;
      u.h[0] = *(const v8bf*)(p0);
      u.h[1] = *(const v8bf*)(p0 + 16);
      bfr[nt] = u.v;
    }

#pragma unroll
    for (int mt = 0; mt < 2; ++mt)
#pragma unroll
      for (int nt = 0; nt < 4; ++nt)
        acc[mt][nt] = __builtin_amdgcn_wmma_f32_16x16x32_bf16(
            false, afr[mt], false, bfr[nt], (short)0, acc[mt][nt], false, false);

    // ---- rotate buffers: refill buf[cur] with tile kt+2, wait for kt+1
    if (kt + 1 < NK) {
      __syncthreads();                 // everyone done reading buf[cur]
      if (wid == 0) {
        if (kt + 2 < NK) {
          const size_t ko = (size_t)(kt + 2) * BK;
          tdm_load_tile(gA + ko, ldsA[cur], (unsigned)K);
          tdm_load_tile(gB + ko, ldsB[cur], (unsigned)K);
          __builtin_amdgcn_s_wait_tensorcnt(2);  // tile kt+1 complete
        } else {
          __builtin_amdgcn_s_wait_tensorcnt(0);
        }
      }
      __syncthreads();                 // buf[1-cur] now valid for all waves
    }
  }

  // ---- epilogue: C layout -> lane L: n = L&15; VGPR g: m = g + (L>>4)*8
#pragma unroll
  for (int mt = 0; mt < 2; ++mt) {
#pragma unroll
    for (int nt = 0; nt < 4; ++nt) {
      const int col = blockN0 + wn * 64 + nt * 16 + (lane & 15);
      const float bv = (EPI & EPI_BIAS) ? bias[col] : 0.0f;
#pragma unroll
      for (int g = 0; g < 8; ++g) {
        const int row = blockM0 + wm * 32 + mt * 16 + (lane >> 4) * 8 + g;
        float v = acc[mt][nt][g] + bv;
        if (EPI & EPI_LEAKY) v = (v > 0.0f) ? v : 0.1f * v;
        if (EPI & EPI_RES)   v += res[(size_t)row * N + col];
        if (EPI & EPI_F32)   Cf[(size_t)row * N + col] = v;
        if (EPI & EPI_BF16)  Cb[(size_t)row * N + col] = f2bf(v);
        if (EPI & EPI_BF16T) Cb[(size_t)col * ldt + row] = f2bf(v);
      }
    }
  }
}

// ---------------------------------------------------------------------------
// Row softmax over f32 scores (scale folded in), emits bf16 weights.
// ---------------------------------------------------------------------------
__global__ __launch_bounds__(256)
void softmax_row_kernel(const float* __restrict__ Sc,
                        bf16_t* __restrict__ W,
                        int n, float scale) {
  const int row = blockIdx.x;
  const int tid = threadIdx.x;
  const float* sr = Sc + (size_t)row * n;
  __shared__ float red[256];

  float m = -3.4e38f;
  for (int i = tid; i < n; i += 256) m = fmaxf(m, sr[i] * scale);
  red[tid] = m; __syncthreads();
  for (int st = 128; st > 0; st >>= 1) {
    if (tid < st) red[tid] = fmaxf(red[tid], red[tid + st]);
    __syncthreads();
  }
  m = red[0]; __syncthreads();

  float sum = 0.f;
  for (int i = tid; i < n; i += 256) sum += __expf(sr[i] * scale - m);
  red[tid] = sum; __syncthreads();
  for (int st = 128; st > 0; st >>= 1) {
    if (tid < st) red[tid] += red[tid + st];
    __syncthreads();
  }
  const float inv = 1.0f / red[0];

  for (int i = tid; i < n; i += 256)
    W[(size_t)row * n + i] = f2bf(__expf(sr[i] * scale - m) * inv);
}

// ---------------------------------------------------------------------------
// Launch. Inputs (f32): x, norm_g, norm_b, Wq, bq, Wk, bk, Wv, bv, W1, b1, W2, b2
// ---------------------------------------------------------------------------
extern "C" void kernel_launch(void* const* d_in, const int* in_sizes, int n_in,
                              void* d_out, int out_size, void* d_ws, size_t ws_size,
                              hipStream_t stream) {
  (void)in_sizes; (void)n_in; (void)out_size; (void)ws_size;
  const float* x      = (const float*)d_in[0];
  const float* norm_g = (const float*)d_in[1];
  const float* norm_b = (const float*)d_in[2];
  const float* Wq = (const float*)d_in[3];  const float* bq = (const float*)d_in[4];
  const float* Wk = (const float*)d_in[5];  const float* bk = (const float*)d_in[6];
  const float* Wv = (const float*)d_in[7];  const float* bv = (const float*)d_in[8];
  const float* W1 = (const float*)d_in[9];  const float* b1 = (const float*)d_in[10];
  const float* W2 = (const float*)d_in[11]; const float* b2 = (const float*)d_in[12];
  float* out = (float*)d_out;

  const int Sd = S_DIM, Nd = N_DIM, Fd = F_DIM;

  // workspace carve-out (~214 MB)
  char* ws = (char*)d_ws;
  size_t off = 0;
  auto take = [&](size_t bytes) -> char* {
    char* p = ws + off;
    off += (bytes + 255) & ~(size_t)255;
    return p;
  };
  float*  xn    = (float*) take((size_t)Sd * Nd * 4);  // normed x (residual)
  bf16_t* xnb   = (bf16_t*)take((size_t)Sd * Nd * 2);
  bf16_t* wqT   = (bf16_t*)take((size_t)Nd * Nd * 2);
  bf16_t* wkT   = (bf16_t*)take((size_t)Nd * Nd * 2);
  bf16_t* wvT   = (bf16_t*)take((size_t)Nd * Nd * 2);
  bf16_t* w1T   = (bf16_t*)take((size_t)Nd * Fd * 2);  // [4N, N]
  bf16_t* w2T   = (bf16_t*)take((size_t)Fd * Nd * 2);  // [N, 4N]
  bf16_t* qb    = (bf16_t*)take((size_t)Sd * Nd * 2);
  bf16_t* kb    = (bf16_t*)take((size_t)Sd * Nd * 2);
  bf16_t* vTb   = (bf16_t*)take((size_t)Nd * Sd * 2);  // v^T
  float*  score = (float*) take((size_t)Sd * Sd * 4);
  bf16_t* wb    = (bf16_t*)take((size_t)Sd * Sd * 2);
  bf16_t* hb    = (bf16_t*)take((size_t)Sd * Nd * 2);
  bf16_t* f1b   = (bf16_t*)take((size_t)Sd * Fd * 2);

  // 1) LayerNorm
  layernorm_kernel<<<Sd, 256, 0, stream>>>(x, norm_g, norm_b, xn, xnb);

  // 2) weight transpose + bf16 convert
  transpose_cvt_kernel<<<dim3(Nd / 32, Nd / 32), 256, 0, stream>>>(Wq, wqT, Nd, Nd);
  transpose_cvt_kernel<<<dim3(Nd / 32, Nd / 32), 256, 0, stream>>>(Wk, wkT, Nd, Nd);
  transpose_cvt_kernel<<<dim3(Nd / 32, Nd / 32), 256, 0, stream>>>(Wv, wvT, Nd, Nd);
  transpose_cvt_kernel<<<dim3(Fd / 32, Nd / 32), 256, 0, stream>>>(W1, w1T, Nd, Fd);
  transpose_cvt_kernel<<<dim3(Nd / 32, Fd / 32), 256, 0, stream>>>(W2, w2T, Fd, Nd);

  // 3) q, k (bf16 row-major); v stored transposed for the w@v GEMM
  gemm_wmma_kernel<EPI_BIAS | EPI_BF16>
      <<<dim3(Nd / 128, Sd / 128), 256, 0, stream>>>(xnb, wqT, bq, nullptr,
                                                     nullptr, qb, Sd, Nd, Nd, 0);
  gemm_wmma_kernel<EPI_BIAS | EPI_BF16>
      <<<dim3(Nd / 128, Sd / 128), 256, 0, stream>>>(xnb, wkT, bk, nullptr,
                                                     nullptr, kb, Sd, Nd, Nd, 0);
  gemm_wmma_kernel<EPI_BIAS | EPI_BF16T>
      <<<dim3(Nd / 128, Sd / 128), 256, 0, stream>>>(xnb, wvT, bv, nullptr,
                                                     nullptr, vTb, Sd, Nd, Nd, Sd);

  // 4) scores = k @ q^T  (q row-major *is* Bt for this convention)
  gemm_wmma_kernel<EPI_F32>
      <<<dim3(Sd / 128, Sd / 128), 256, 0, stream>>>(kb, qb, nullptr, nullptr,
                                                     score, nullptr, Sd, Sd, Nd, 0);

  // 5) softmax rows, scale = 1/sqrt(1024)
  softmax_row_kernel<<<Sd, 256, 0, stream>>>(score, wb, Sd, 0.03125f);

  // 6) h = w @ v
  gemm_wmma_kernel<EPI_BF16>
      <<<dim3(Nd / 128, Sd / 128), 256, 0, stream>>>(wb, vTb, nullptr, nullptr,
                                                     nullptr, hb, Sd, Nd, Sd, 0);

  // 7) f1 = leaky_relu(h @ W1 + b1)
  gemm_wmma_kernel<EPI_BIAS | EPI_LEAKY | EPI_BF16>
      <<<dim3(Fd / 128, Sd / 128), 256, 0, stream>>>(hb, w1T, b1, nullptr,
                                                     nullptr, f1b, Sd, Fd, Nd, 0);

  // 8) out = f1 @ W2 + b2 + xn
  gemm_wmma_kernel<EPI_BIAS | EPI_RES | EPI_F32>
      <<<dim3(Nd / 128, Sd / 128), 256, 0, stream>>>(f1b, w2T, b2, xn,
                                                     out, nullptr, Sd, Nd, Fd, 0);
}
